// BiCrossAttention_38749194945183
// MI455X (gfx1250) — matrix operand order repaired and checked by reading, not compiled, not run on hardware
//
#include <hip/hip_runtime.h>

// ---------------------------------------------------------------------------
// BiCrossAttention for MI455X (gfx1250): bf16 WMMA GEMMs + double-buffered
// TDM flash attention.
// ---------------------------------------------------------------------------

typedef __attribute__((ext_vector_type(8)))  __bf16        v8bf;
typedef __attribute__((ext_vector_type(16))) __bf16        v16bf;
typedef __attribute__((ext_vector_type(8)))  float         v8f;
typedef __attribute__((ext_vector_type(4)))  unsigned int  v4u;
typedef __attribute__((ext_vector_type(8)))  int           v8i;
typedef __attribute__((ext_vector_type(4)))  int           v4i;

#define BB    2
#define SEQ   2048
#define QDIM  1024
#define HH    8
#define DHD   64
#define INNR  512          // H * DH
#define SCL   0.125f       // DH^-0.5

// ---------------------------------------------------------------------------
// WMMA helpers
// ---------------------------------------------------------------------------
__device__ __forceinline__ v8f wmma_bf16(v16bf a, v16bf b, v8f c) {
  return __builtin_amdgcn_wmma_f32_16x16x32_bf16(
      false, a, false, b, (short)0, c, false, false);
}

// A-operand fragment, 16x32 bf16 (rows r0..r0+15, K k0..k0+31).
__device__ __forceinline__ v16bf load_a_frag(const __bf16* base, int ld,
                                             int r0, int k0) {
  int lane = threadIdx.x & 31;
  int r  = r0 + (lane & 15);
  int kb = k0 + ((lane >> 4) << 3);
  const __bf16* p = base + (size_t)r * ld + kb;
  v8bf lo = *(const v8bf*)(p);
  v8bf hi = *(const v8bf*)(p + 16);
  v16bf out;
#pragma unroll
  for (int i = 0; i < 8; ++i) { out[i] = lo[i]; out[i + 8] = hi[i]; }
  return out;
}

// B-operand fragment, 32x16 bf16 given B^T row-major [Ncols x K].
__device__ __forceinline__ v16bf load_b_frag(const __bf16* baseT, int ld,
                                             int n0, int k0) {
  int lane = threadIdx.x & 31;
  int n  = n0 + (lane & 15);
  int kb = k0 + ((lane >> 4) << 3);
  const __bf16* p = baseT + (size_t)n * ld + kb;
  v8bf lo = *(const v8bf*)(p);
  v8bf hi = *(const v8bf*)(p + 16);
  v16bf out;
#pragma unroll
  for (int i = 0; i < 8; ++i) { out[i] = lo[i]; out[i + 8] = hi[i]; }
  return out;
}

// ---------------------------------------------------------------------------
// Tensor Data Mover: async 2D tile load (global -> LDS), bf16 elements.
// D# per CDNA5 ISA. Tensor == tile (no OOB needed),
// tile_d0 contiguous, rows strided by stride0 elements.
// ---------------------------------------------------------------------------
__device__ __forceinline__ void tdm_load_2d(unsigned int lds_off,
                                            const void* gptr,
                                            unsigned int tile_d0,
                                            unsigned int tile_d1,
                                            unsigned long long stride0) {
  unsigned long long ga = (unsigned long long)(uintptr_t)gptr;
  v4u g0;
  g0[0] = 1u;                                   // count=1, user mode
  g0[1] = lds_off;                              // lds_addr (bytes)
  g0[2] = (unsigned int)ga;                     // global_addr[31:0]
  g0[3] = (unsigned int)((ga >> 32) & 0x1FFFFFFu) | (2u << 30); // [56:32]+type=2
  v8i g1;
  g1[0] = (int)(1u << 16);                      // wg_mask=0, data_size=1 (2B)
  g1[1] = (int)((tile_d0 & 0xFFFFu) << 16);     // tensor_dim0[15:0]
  g1[2] = (int)(((tile_d0 >> 16) & 0xFFFFu) | ((tile_d1 & 0xFFFFu) << 16));
  g1[3] = (int)(((tile_d1 >> 16) & 0xFFFFu) | ((tile_d0 & 0xFFFFu) << 16)); // tile_dim0
  g1[4] = (int)(tile_d1 & 0xFFFFu);             // tile_dim1, tile_dim2=0
  g1[5] = (int)(unsigned int)(stride0 & 0xFFFFFFFFu);       // dim0_stride[31:0]
  g1[6] = (int)(unsigned int)((stride0 >> 32) & 0xFFFFu);   // dim0_stride[47:32]
  g1[7] = 0;
  v4i gz = {0, 0, 0, 0};
#if defined(__clang_major__) && (__clang_major__ >= 23)
  v8i gz8 = {0, 0, 0, 0, 0, 0, 0, 0};
  __builtin_amdgcn_tensor_load_to_lds(g0, g1, gz, gz, gz8, 0);
#else
  __builtin_amdgcn_tensor_load_to_lds(g0, g1, gz, gz, 0);
#endif
}

// ---------------------------------------------------------------------------
// Elementwise converters
// ---------------------------------------------------------------------------
__global__ void __launch_bounds__(256) cvt_f32_bf16(const float* __restrict__ in,
                                                    __bf16* __restrict__ out,
                                                    int n) {
  int i = blockIdx.x * 256 + threadIdx.x;
  if (i < n) out[i] = (__bf16)in[i];
}

// W [K x N] fp32 -> WT [N x K] bf16
__global__ void __launch_bounds__(256) trcvt_f32_bf16(const float* __restrict__ W,
                                                      __bf16* __restrict__ WT,
                                                      int K, int N) {
  int i = blockIdx.x * 256 + threadIdx.x;
  if (i < K * N) {
    int k = i / N, n = i - k * N;
    WT[(size_t)n * K + k] = (__bf16)W[i];
  }
}

// out_bf16 = alpha*o1 + (1-alpha)*o2, alpha = sigmoid(*alpha_p)
__global__ void __launch_bounds__(256) blend_bf16(const float* __restrict__ o1,
                                                  const float* __restrict__ o2,
                                                  const float* __restrict__ alpha_p,
                                                  __bf16* __restrict__ out, int n) {
  int i = blockIdx.x * 256 + threadIdx.x;
  if (i < n) {
    float a = 1.0f / (1.0f + __expf(-alpha_p[0]));
    out[i] = (__bf16)(o1[i] * a + o2[i] * (1.0f - a));
  }
}

// ---------------------------------------------------------------------------
// Projection GEMM: C[4096 x 512] = A[4096 x 1024] x W (given W^T bf16)
// 32x64 register tile per wave (2 A-frags x 4 B-frags = 8 wmma / k-step).
// mode 0: store [B,H,S,D]   mode 1: store V transposed [B,H,D,S]
// ---------------------------------------------------------------------------
__global__ void __launch_bounds__(256) gemm_proj(const __bf16* __restrict__ A,
                                                 const __bf16* __restrict__ WT,
                                                 __bf16* __restrict__ out,
                                                 int mode) {
  int wave = blockIdx.x * 8 + (threadIdx.x >> 5);
  int lane = threadIdx.x & 31;
  const int NT = INNR / 64;               // 8 col blocks
  int nt = wave % NT, mt = wave / NT;     // mt in [0,128)
  int m0 = mt * 32, n0 = nt * 64;
  v8f acc[2][4] = {};
#pragma unroll 2
  for (int k0 = 0; k0 < QDIM; k0 += 32) {
    __builtin_prefetch(A + (size_t)(m0 + (lane & 15)) * QDIM + k0 + 32, 0, 0);
    v16bf a0 = load_a_frag(A, QDIM, m0, k0);
    v16bf a1 = load_a_frag(A, QDIM, m0 + 16, k0);
    v16bf b[4];
#pragma unroll
    for (int j = 0; j < 4; ++j) b[j] = load_b_frag(WT, QDIM, n0 + 16 * j, k0);
#pragma unroll
    for (int j = 0; j < 4; ++j) {
      acc[0][j] = wmma_bf16(a0, b[j], acc[0][j]);
      acc[1][j] = wmma_bf16(a1, b[j], acc[1][j]);
    }
  }
  int roff = (lane >> 4) << 3;
#pragma unroll
  for (int i = 0; i < 2; ++i) {
    int rbase = m0 + 16 * i + roff;
#pragma unroll
    for (int j = 0; j < 4; ++j) {
      int col = n0 + 16 * j + (lane & 15);
      int h = col >> 6, d = col & 63;
#pragma unroll
      for (int r = 0; r < 8; ++r) {
        int gr = rbase + r;
        int b2 = gr >> 11, s = gr & (SEQ - 1);
        size_t idx;
        if (mode == 0) idx = (((size_t)b2 * HH + h) * SEQ + s) * DHD + d;
        else           idx = (((size_t)b2 * HH + h) * DHD + d) * SEQ + s;
        out[idx] = (__bf16)acc[i][j][r];
      }
    }
  }
}

// ---------------------------------------------------------------------------
// Flash attention: softmax(q k^T * scale) v   (one 16-row q tile per wave)
// 4 waves/block share one (b,h); K/V j-chunks double-buffered in LDS via the
// Tensor Data Mover so the DMA for chunk j+1 overlaps compute on chunk j.
// q,k : bf16 [B,H,S,D]     vT : bf16 [B,H,D,S]     out : f32 [B,S,INNER]
// ---------------------------------------------------------------------------
__global__ void __launch_bounds__(128) flash_attn(const __bf16* __restrict__ q,
                                                  const __bf16* __restrict__ k,
                                                  const __bf16* __restrict__ vT,
                                                  float* __restrict__ out) {
  __shared__ __bf16 ldsK[2][32 * 64];     // [buf][j_local][d]   2 x 4 KB
  __shared__ __bf16 ldsV[2][64 * 32];     // [buf][d][j_local]   2 x 4 KB
  __shared__ __bf16 ldsP[4][16 * 32];     // per-wave P staging (1 KB each)
  int wv   = threadIdx.x >> 5;
  int lane = threadIdx.x & 31;
  int gw   = blockIdx.x * 4 + wv;         // [0, B*H*S/16) = 2048
  int st   = gw & (SEQ / 16 - 1);
  int bh   = gw / (SEQ / 16);             // 4 waves/block: same bh (128 % 4 == 0)
  int b    = bh / HH, h = bh - b * HH;
  const __bf16* qb = q  + (size_t)bh * SEQ * DHD;
  const __bf16* kb = k  + (size_t)bh * SEQ * DHD;
  const __bf16* vb = vT + (size_t)bh * DHD * SEQ;
  int row0 = st * 16;

  v16bf qf0 = load_a_frag(qb, DHD, row0, 0);
  v16bf qf1 = load_a_frag(qb, DHD, row0, 32);

  v8f acc[4] = {};
  float m[8], l[8];
#pragma unroll
  for (int r = 0; r < 8; ++r) { m[r] = -__builtin_inff(); l[r] = 0.0f; }

  __bf16* pt = &ldsP[wv][0];
  unsigned int kOff[2] = { (unsigned int)(uintptr_t)(void*)&ldsK[0][0],
                           (unsigned int)(uintptr_t)(void*)&ldsK[1][0] };
  unsigned int vOff[2] = { (unsigned int)(uintptr_t)(void*)&ldsV[0][0],
                           (unsigned int)(uintptr_t)(void*)&ldsV[1][0] };

  // prologue: start DMA for chunk 0
  if (wv == 0) {
    tdm_load_2d(kOff[0], kb, 64u, 32u, (unsigned long long)DHD);
    tdm_load_2d(vOff[0], vb, 32u, 64u, (unsigned long long)SEQ);
  }

  for (int j0 = 0; j0 < SEQ; j0 += 32) {
    int cur = (j0 >> 5) & 1;
    __syncthreads();  // all waves done reading buffer cur^1 (iteration j0-32)
    if (wv == 0) {
      if (j0 + 32 < SEQ) {
        // issue DMA for next chunk into the other buffer, then wait only for
        // the current chunk (TDM ops retire in order: <=2 outstanding means
        // chunk j0's two loads have landed).
        tdm_load_2d(kOff[cur ^ 1], kb + (size_t)(j0 + 32) * DHD, 64u, 32u,
                    (unsigned long long)DHD);
        tdm_load_2d(vOff[cur ^ 1], vb + (j0 + 32), 32u, 64u,
                    (unsigned long long)SEQ);
        __builtin_amdgcn_s_wait_tensorcnt(2);
      } else {
        __builtin_amdgcn_s_wait_tensorcnt(0);
      }
    }
    __syncthreads();  // buffer cur visible to all waves
    const __bf16* Kc = &ldsK[cur][0];
    const __bf16* Vc = &ldsV[cur][0];

    // scores: two 16x16 tiles from LDS K, K-dim = DHD = 64 (2 wmma each)
    v8f c0 = {}, c1 = {};
    c0 = wmma_bf16(qf0, load_b_frag(Kc, DHD, 0, 0), c0);
    c0 = wmma_bf16(qf1, load_b_frag(Kc, DHD, 0, 32), c0);
    c1 = wmma_bf16(qf0, load_b_frag(Kc, DHD, 16, 0), c1);
    c1 = wmma_bf16(qf1, load_b_frag(Kc, DHD, 16, 32), c1);

    float s0[8], s1[8], t[8];
#pragma unroll
    for (int r = 0; r < 8; ++r) {
      s0[r] = c0[r] * SCL;
      s1[r] = c1[r] * SCL;
      t[r]  = fmaxf(s0[r], s1[r]);
    }
#pragma unroll
    for (int r = 0; r < 8; ++r) {
#pragma unroll
      for (int msk = 1; msk < 16; msk <<= 1)
        t[r] = fmaxf(t[r], __shfl_xor(t[r], msk, 32));
    }
    float f[8], rs[8];
#pragma unroll
    for (int r = 0; r < 8; ++r) {
      float mn = fmaxf(m[r], t[r]);
      f[r]  = __expf(m[r] - mn);
      m[r]  = mn;
      s0[r] = __expf(s0[r] - mn);
      s1[r] = __expf(s1[r] - mn);
      rs[r] = s0[r] + s1[r];
    }
#pragma unroll
    for (int r = 0; r < 8; ++r) {
#pragma unroll
      for (int msk = 1; msk < 16; msk <<= 1)
        rs[r] += __shfl_xor(rs[r], msk, 32);
      l[r] = l[r] * f[r] + rs[r];
    }
#pragma unroll
    for (int tt = 0; tt < 4; ++tt)
#pragma unroll
      for (int r = 0; r < 8; ++r) acc[tt][r] *= f[r];

    // C-layout -> A-fragment layout through per-wave LDS tile
    int prow = (lane >> 4) << 3;
    int pcol = lane & 15;
#pragma unroll
    for (int r = 0; r < 8; ++r) {
      pt[(prow + r) * 32 + pcol]      = (__bf16)s0[r];
      pt[(prow + r) * 32 + 16 + pcol] = (__bf16)s1[r];
    }
    __syncthreads();                      // also fences same-wave LDS RAW
    v16bf pf = load_a_frag(pt, 32, 0, 0);

    // acc += P @ V  (V^T rows contiguous along local j in LDS)
#pragma unroll
    for (int tt = 0; tt < 4; ++tt)
      acc[tt] = wmma_bf16(pf, load_b_frag(Vc, 32, tt * 16, 0), acc[tt]);
  }

  // epilogue: out[b, s, h*DH + d] = acc / l
  float* ob = out + (size_t)b * SEQ * INNR + (size_t)h * DHD;
  int col  = lane & 15;
  int roff = (lane >> 4) << 3;
#pragma unroll
  for (int r = 0; r < 8; ++r) {
    float inv = 1.0f / l[r];
    int s = row0 + roff + r;
#pragma unroll
    for (int tt = 0; tt < 4; ++tt)
      ob[(size_t)s * INNR + tt * 16 + col] = acc[tt][r] * inv;
  }
}

// ---------------------------------------------------------------------------
// Output GEMM: out[4096 x 1024] = Ab[4096 x 512] x Wout (W^T bf16) + bias
// 32x64 register tile per wave.
// ---------------------------------------------------------------------------
__global__ void __launch_bounds__(256) gemm_out(const __bf16* __restrict__ A,
                                                const __bf16* __restrict__ WT,
                                                const float* __restrict__ bias,
                                                float* __restrict__ out) {
  int wave = blockIdx.x * 8 + (threadIdx.x >> 5);
  int lane = threadIdx.x & 31;
  const int NT = QDIM / 64;               // 16 col blocks
  int nt = wave % NT, mt = wave / NT;     // mt in [0,128)
  int m0 = mt * 32, n0 = nt * 64;
  v8f acc[2][4] = {};
#pragma unroll 2
  for (int k0 = 0; k0 < INNR; k0 += 32) {
    __builtin_prefetch(A + (size_t)(m0 + (lane & 15)) * INNR + k0 + 32, 0, 0);
    v16bf a0 = load_a_frag(A, INNR, m0, k0);
    v16bf a1 = load_a_frag(A, INNR, m0 + 16, k0);
    v16bf b[4];
#pragma unroll
    for (int j = 0; j < 4; ++j) b[j] = load_b_frag(WT, INNR, n0 + 16 * j, k0);
#pragma unroll
    for (int j = 0; j < 4; ++j) {
      acc[0][j] = wmma_bf16(a0, b[j], acc[0][j]);
      acc[1][j] = wmma_bf16(a1, b[j], acc[1][j]);
    }
  }
  int roff = (lane >> 4) << 3;
#pragma unroll
  for (int i = 0; i < 2; ++i) {
    int rbase = m0 + 16 * i + roff;
#pragma unroll
    for (int j = 0; j < 4; ++j) {
      int col = n0 + 16 * j + (lane & 15);
      float bv = bias[col];
#pragma unroll
      for (int r = 0; r < 8; ++r)
        out[(size_t)(rbase + r) * QDIM + col] = acc[i][j][r] + bv;
    }
  }
}

// ---------------------------------------------------------------------------
// Host launcher
// ---------------------------------------------------------------------------
extern "C" void kernel_launch(void* const* d_in, const int* in_sizes, int n_in,
                              void* d_out, int out_size, void* d_ws, size_t ws_size,
                              hipStream_t stream) {
  (void)in_sizes; (void)n_in; (void)out_size; (void)ws_size;
  const float* x     = (const float*)d_in[0];
  const float* ctx   = (const float*)d_in[1];
  const float* Wq1   = (const float*)d_in[2];
  const float* Wk1   = (const float*)d_in[3];
  const float* Wv1   = (const float*)d_in[4];
  const float* Wq2   = (const float*)d_in[5];
  const float* Wk2   = (const float*)d_in[6];
  const float* Wv2   = (const float*)d_in[7];
  const float* alpha = (const float*)d_in[8];
  const float* Wout  = (const float*)d_in[9];
  const float* bout  = (const float*)d_in[10];
  float* out = (float*)d_out;

  char* ws = (char*)d_ws;
  size_t off = 0;
  auto salloc = [&](size_t bytes) -> void* {
    void* p = ws + off;
    off += (bytes + 255) & ~(size_t)255;
    return p;
  };
  const size_t NX = (size_t)BB * SEQ * QDIM;
  const size_t NW = (size_t)QDIM * INNR;
  const size_t NH = (size_t)BB * HH * SEQ * DHD;
  const size_t NO = (size_t)BB * SEQ * INNR;

  __bf16* xb   = (__bf16*)salloc(NX * 2);
  __bf16* cb   = (__bf16*)salloc(NX * 2);
  __bf16* wq1t = (__bf16*)salloc(NW * 2);
  __bf16* wk1t = (__bf16*)salloc(NW * 2);
  __bf16* wv1t = (__bf16*)salloc(NW * 2);
  __bf16* wq2t = (__bf16*)salloc(NW * 2);
  __bf16* wk2t = (__bf16*)salloc(NW * 2);
  __bf16* wv2t = (__bf16*)salloc(NW * 2);
  __bf16* woT  = (__bf16*)salloc(NW * 2);
  __bf16* q1   = (__bf16*)salloc(NH * 2);
  __bf16* k1   = (__bf16*)salloc(NH * 2);
  __bf16* v1T  = (__bf16*)salloc(NH * 2);
  __bf16* q2   = (__bf16*)salloc(NH * 2);
  __bf16* k2   = (__bf16*)salloc(NH * 2);
  __bf16* v2T  = (__bf16*)salloc(NH * 2);
  float*  o1   = (float*)salloc(NO * 4);
  float*  o2   = (float*)salloc(NO * 4);
  __bf16* obld = (__bf16*)salloc(NO * 2);

  cvt_f32_bf16<<<(int)((NX + 255) / 256), 256, 0, stream>>>(x, xb, (int)NX);
  cvt_f32_bf16<<<(int)((NX + 255) / 256), 256, 0, stream>>>(ctx, cb, (int)NX);
  int wgrid = (int)((NW + 255) / 256);
  trcvt_f32_bf16<<<wgrid, 256, 0, stream>>>(Wq1, wq1t, QDIM, INNR);
  trcvt_f32_bf16<<<wgrid, 256, 0, stream>>>(Wk1, wk1t, QDIM, INNR);
  trcvt_f32_bf16<<<wgrid, 256, 0, stream>>>(Wv1, wv1t, QDIM, INNR);
  trcvt_f32_bf16<<<wgrid, 256, 0, stream>>>(Wq2, wq2t, QDIM, INNR);
  trcvt_f32_bf16<<<wgrid, 256, 0, stream>>>(Wk2, wk2t, QDIM, INNR);
  trcvt_f32_bf16<<<wgrid, 256, 0, stream>>>(Wv2, wv2t, QDIM, INNR);
  trcvt_f32_bf16<<<wgrid, 256, 0, stream>>>(Wout, woT, INNR, QDIM);

  // projections: (4096/32)*(512/64) = 1024 waves = 128 blocks x 8 waves
  gemm_proj<<<128, 256, 0, stream>>>(xb, wq1t, q1, 0);
  gemm_proj<<<128, 256, 0, stream>>>(xb, wk1t, k1, 0);
  gemm_proj<<<128, 256, 0, stream>>>(xb, wv1t, v1T, 1);
  gemm_proj<<<128, 256, 0, stream>>>(cb, wq2t, q2, 0);
  gemm_proj<<<128, 256, 0, stream>>>(cb, wk2t, k2, 0);
  gemm_proj<<<128, 256, 0, stream>>>(cb, wv2t, v2T, 1);

  // flash attention, both directions: 2048 waves = 512 blocks x 128 threads
  flash_attn<<<512, 128, 0, stream>>>(q2, k1, v1T, o1);   // out1
  flash_attn<<<512, 128, 0, stream>>>(q1, k2, v2T, o2);   // out2

  blend_bf16<<<(int)((NO + 255) / 256), 256, 0, stream>>>(o1, o2, alpha, obld, (int)NO);

  // output projection: (4096/32)*(1024/64) = 2048 waves = 256 blocks
  gemm_out<<<256, 256, 0, stream>>>(obld, woT, bout, out);
}
